// VisionMambaEncoderBlock_40037685134204
// MI455X (gfx1250) — matrix operand order repaired and checked by reading, not compiled, not run on hardware
//
#include <hip/hip_runtime.h>
#include <hip/hip_bf16.h>

typedef _Float16 half_t;
typedef __attribute__((ext_vector_type(16))) _Float16 v16h;
typedef __attribute__((ext_vector_type(8)))  _Float16 v8h;
typedef __attribute__((ext_vector_type(8)))  float    v8f;

#define D_MODEL 384
#define D_STATE 16
#define D_CONV  4
#define D_INNER 768
#define DT_RANK 24
#define BATCH   4
#define SEQ     1024
#define TOK     (BATCH*SEQ)   // 4096 tokens

__device__ __forceinline__ float silu_f(float x) { return x / (1.f + __expf(-x)); }
__device__ __forceinline__ float softplus_f(float x) {
  return (x > 20.f) ? x : log1pf(__expf(x));
}

// ---------------------------------------------------------------------------
// LayerNorm: one block per token, writes f16 normalized activations.
// ---------------------------------------------------------------------------
__global__ void ln_kernel(const float* __restrict__ x, const float* __restrict__ g,
                          const float* __restrict__ bta, half_t* __restrict__ xnH)
{
  __shared__ float s1[128], s2[128];
  const int tok = blockIdx.x;
  const float* row = x + (size_t)tok * D_MODEL;
  float a = 0.f, b = 0.f;
  for (int c = threadIdx.x; c < D_MODEL; c += 128) { float v = row[c]; a += v; b += v * v; }
  s1[threadIdx.x] = a; s2[threadIdx.x] = b;
  __syncthreads();
  for (int st = 64; st > 0; st >>= 1) {
    if ((int)threadIdx.x < st) { s1[threadIdx.x] += s1[threadIdx.x + st]; s2[threadIdx.x] += s2[threadIdx.x + st]; }
    __syncthreads();
  }
  const float mu  = s1[0] / (float)D_MODEL;
  const float var = s2[0] / (float)D_MODEL - mu * mu;
  const float rs  = rsqrtf(var + 1e-5f);
  for (int c = threadIdx.x; c < D_MODEL; c += 128) {
    float v = (row[c] - mu) * rs * g[c] + bta[c];
    xnH[(size_t)tok * D_MODEL + c] = (half_t)v;
  }
}

// ---------------------------------------------------------------------------
// f32 -> f16 converters (weights are tiny; this is bandwidth-trivial).
// ---------------------------------------------------------------------------
__global__ void cvt_f16_kernel(const float* __restrict__ s, half_t* __restrict__ d, int n)
{
  int i = blockIdx.x * 256 + threadIdx.x;
  if (i < n) d[i] = (half_t)s[i];
}

__global__ void cvt_pad_kernel(const float* __restrict__ s, half_t* __restrict__ d,
                               int Ns, int Ks, int Nd, int Kd)
{
  int i = blockIdx.x * 256 + threadIdx.x;
  if (i >= Nd * Kd) return;
  int r = i / Kd, c = i % Kd;
  float v = (r < Ns && c < Ks) ? s[r * Ks + c] : 0.f;
  d[i] = (half_t)v;
}

// ---------------------------------------------------------------------------
// WMMA fragment loads (wave32 layouts per CDNA5 ISA 7.12.2).
// A 16x32 f16: lane m=lane&15; halves 0..7 = K[base..base+7], 8..15 = K[base+16..+23],
//              base = (lane>=16) ? 8 : 0  -> two 16B loads with a 16-half gap.
// B 32x16 f16: lane n=lane&15; halves 0..7 = K[base..base+7], 8..15 = K[base+16..+23], same as A.
// C/D 16x16 f32: lane col n=lane&15; vgpr j = row (lane>=16)*8 + j.
// ---------------------------------------------------------------------------
__device__ __forceinline__ v16h ldfrag_gap(const half_t* __restrict__ p)
{
  v8h lo = *(const v8h*)p;
  v8h hi = *(const v8h*)(p + 16);
  v16h r;
#pragma unroll
  for (int e = 0; e < 8; ++e) { r[e] = lo[e]; r[8 + e] = hi[e]; }
  return r;
}

__device__ __forceinline__ v16h ldfrag_ctg(const half_t* __restrict__ p)
{
  v8h lo = *(const v8h*)p;
  v8h hi = *(const v8h*)(p + 8);
  v16h r;
#pragma unroll
  for (int e = 0; e < 8; ++e) { r[e] = lo[e]; r[8 + e] = hi[e]; }
  return r;
}

// ---------------------------------------------------------------------------
// Generic WMMA GEMM: out[M=4096, N] = act( A[M,K](f16) x W[N,K](f16)^T + bias ).
// One wave computes a 64x16 tile (4 WMMAs per 32-wide k-step, B frag reused 4x).
// act: 0 = none, 1 = silu, 2 = softplus.
// ---------------------------------------------------------------------------
__global__ void wmma_gemm(const half_t* __restrict__ A, const half_t* __restrict__ W,
                          const float* __restrict__ bias, float* __restrict__ outF,
                          half_t* __restrict__ outH, int N, int K, int act)
{
  const int lane = threadIdx.x & 31;
  const int wave = threadIdx.x >> 5;
  const int mGroups = TOK >> 6;       // 64 rows per task
  const int nTiles  = N >> 4;
  const int task = blockIdx.x * 8 + wave;
  if (task >= mGroups * nTiles) return;   // whole-wave uniform exit: EXEC stays all-1s

  const int m0    = (task % mGroups) << 6;
  const int ncol  = ((task / mGroups) << 4) + (lane & 15);
  const int abase = (lane >> 4) << 3;     // 0 or 8
  const int koff  = (lane >> 4) << 4;     // 0 or 16

  const half_t* wrow  = W + (size_t)ncol * K + abase;
  const half_t* arow0 = A + (size_t)(m0 + (lane & 15)) * K + abase;
  const size_t rstride = (size_t)16 * K;

  v8f acc0 = {}, acc1 = {}, acc2 = {}, acc3 = {};
  for (int k = 0; k < K; k += 32) {
    v16h bf = ldfrag_gap(wrow + k);
    v16h a0 = ldfrag_gap(arow0 + k);
    v16h a1 = ldfrag_gap(arow0 + rstride + k);
    v16h a2 = ldfrag_gap(arow0 + 2 * rstride + k);
    v16h a3 = ldfrag_gap(arow0 + 3 * rstride + k);
    __builtin_prefetch(wrow + k + 32, 0, 3);
    __builtin_prefetch(arow0 + k + 32, 0, 3);
    acc0 = __builtin_amdgcn_wmma_f32_16x16x32_f16(false, a0, false, bf, (short)0, acc0, false, false);
    acc1 = __builtin_amdgcn_wmma_f32_16x16x32_f16(false, a1, false, bf, (short)0, acc1, false, false);
    acc2 = __builtin_amdgcn_wmma_f32_16x16x32_f16(false, a2, false, bf, (short)0, acc2, false, false);
    acc3 = __builtin_amdgcn_wmma_f32_16x16x32_f16(false, a3, false, bf, (short)0, acc3, false, false);
  }

  const float bv = bias ? bias[ncol] : 0.f;
  const int mb = m0 + ((lane >> 4) << 3);
  auto emit = [&](v8f acc, int mstart) {
#pragma unroll
    for (int j = 0; j < 8; ++j) {
      float v = acc[j] + bv;
      if (act == 1)      v = silu_f(v);
      else if (act == 2) v = softplus_f(v);
      size_t o = (size_t)(mstart + j) * N + ncol;
      if (outF) outF[o] = v;
      if (outH) outH[o] = (half_t)v;
    }
  };
  emit(acc0, mb);
  emit(acc1, mb + 16);
  emit(acc2, mb + 32);
  emit(acc3, mb + 48);
}

// ---------------------------------------------------------------------------
// Depthwise causal conv (K=4) + SiLU over the first D_INNER cols of xz.
// ---------------------------------------------------------------------------
__global__ void conv_silu_kernel(const float* __restrict__ xz, const float* __restrict__ cw,
                                 const float* __restrict__ cb, float* __restrict__ xcF,
                                 half_t* __restrict__ xcH)
{
  int i = blockIdx.x * 256 + threadIdx.x;
  if (i >= TOK * D_INNER) return;
  int d = i % D_INNER;
  int t = (i / D_INNER) % SEQ;
  int b = i / (D_INNER * SEQ);
  float acc = cb[d];
#pragma unroll
  for (int k = 0; k < D_CONV; ++k) {
    int tt = t + k - (D_CONV - 1);
    if (tt >= 0)
      acc += cw[d * D_CONV + k] * xz[(size_t)(b * SEQ + tt) * (2 * D_INNER) + d];
  }
  acc = silu_f(acc);
  xcF[i] = acc;
  xcH[i] = (half_t)acc;
}

// ---------------------------------------------------------------------------
// Split dbc[4096,64] into dtr (f16, K-padded to 32), Bm, Cm.
// ---------------------------------------------------------------------------
__global__ void pack_dbc_kernel(const float* __restrict__ dbc, half_t* __restrict__ dtr,
                                float* __restrict__ Bm, float* __restrict__ Cm)
{
  int i = blockIdx.x * 256 + threadIdx.x;
  if (i >= TOK * 64) return;
  int r = i >> 6, c = i & 63;
  float v = dbc[i];
  if (c < 32)              dtr[r * 32 + c]        = (c < DT_RANK) ? (half_t)v : (half_t)0.f;
  if (c >= 24 && c < 40)   Bm[r * D_STATE + c - 24] = v;
  if (c >= 40 && c < 56)   Cm[r * D_STATE + c - 40] = v;
}

// ---------------------------------------------------------------------------
// Selective scan: one thread per (batch, channel), 16 states in registers,
// per-step B/C broadcast via double-buffered LDS (one barrier per step).
// Fuses  y = (scan + Dp*xc) * silu(z_inner)  and emits f16 for the out-GEMM.
// ---------------------------------------------------------------------------
__global__ void scan_kernel(const float* __restrict__ dt, const float* __restrict__ xc,
                            const float* __restrict__ xz, const float* __restrict__ Bm,
                            const float* __restrict__ Cm, const float* __restrict__ A_log,
                            const float* __restrict__ Dp, half_t* __restrict__ yH)
{
  const int b = blockIdx.x / (D_INNER / 128);
  const int d = (blockIdx.x % (D_INNER / 128)) * 128 + threadIdx.x;

  float Av[D_STATE], h[D_STATE];
#pragma unroll
  for (int n = 0; n < D_STATE; ++n) {
    Av[n] = -__expf(A_log[(size_t)d * D_STATE + n]);
    h[n] = 0.f;
  }
  const float Dv = Dp[d];

  __shared__ float sB[2][D_STATE], sC[2][D_STATE];

  for (int t = 0; t < SEQ; ++t) {
    const int buf = t & 1;
    const size_t tokoff = (size_t)(b * SEQ + t);
    if ((int)threadIdx.x < D_STATE)
      sB[buf][threadIdx.x] = Bm[tokoff * D_STATE + threadIdx.x];
    else if ((int)threadIdx.x < 2 * D_STATE)
      sC[buf][threadIdx.x - D_STATE] = Cm[tokoff * D_STATE + threadIdx.x - D_STATE];
    __syncthreads();

    const float dtv = dt[tokoff * D_INNER + d];
    const float xv  = xc[tokoff * D_INNER + d];
    const float zv  = xz[tokoff * (2 * D_INNER) + D_INNER + d];
    const float dux = dtv * xv;
    float y = 0.f;
#pragma unroll
    for (int n = 0; n < D_STATE; ++n) {
      float dA = __expf(dtv * Av[n]);
      h[n] = dA * h[n] + dux * sB[buf][n];
      y += h[n] * sC[buf][n];
    }
    y = (y + Dv * xv) * silu_f(zv);
    yH[tokoff * D_INNER + d] = (half_t)y;
  }
}

// ---------------------------------------------------------------------------
// Final combine: out = (y0 + y1) * z + skip
// ---------------------------------------------------------------------------
__global__ void final_kernel(const float* __restrict__ y0, const float* __restrict__ y1,
                             const float* __restrict__ z, const float* __restrict__ x,
                             float* __restrict__ out, int n)
{
  int i = blockIdx.x * 256 + threadIdx.x;
  if (i < n) out[i] = (y0[i] + y1[i]) * z[i] + x[i];
}

// ---------------------------------------------------------------------------
extern "C" void kernel_launch(void* const* d_in, const int* in_sizes, int n_in,
                              void* d_out, int out_size, void* d_ws, size_t ws_size,
                              hipStream_t stream)
{
  (void)in_sizes; (void)n_in; (void)out_size; (void)ws_size;

  const float* x        = (const float*)d_in[0];
  const float* ln_g     = (const float*)d_in[1];
  const float* ln_b     = (const float*)d_in[2];
  const float* lin_w    = (const float*)d_in[3];
  const float* lin_b    = (const float*)d_in[4];
  const float* pconv_w  = (const float*)d_in[5];
  const float* pconv_b  = (const float*)d_in[6];
  const float* in_w     = (const float*)d_in[7];
  const float* conv_w   = (const float*)d_in[8];
  const float* conv_b   = (const float*)d_in[9];
  const float* xproj_w  = (const float*)d_in[10];
  const float* dtproj_w = (const float*)d_in[11];
  const float* dtproj_b = (const float*)d_in[12];
  const float* A_log    = (const float*)d_in[13];
  const float* Dp       = (const float*)d_in[14];
  const float* out_w    = (const float*)d_in[15];
  float* out = (float*)d_out;

  char* wsb = (char*)d_ws;
  size_t off = 0;
  auto take = [&](size_t bytes) -> void* {
    void* p = wsb + off;
    off += (bytes + 255) & ~(size_t)255;
    return p;
  };

  // f16 weight copies
  half_t* lin_h    = (half_t*)take((size_t)D_MODEL * D_MODEL * 2);
  half_t* pconv_h  = (half_t*)take((size_t)2 * D_MODEL * D_MODEL * 2);
  half_t* in_h     = (half_t*)take((size_t)2 * (2 * D_INNER) * D_MODEL * 2);
  half_t* xproj_h  = (half_t*)take((size_t)2 * 64 * D_INNER * 2);      // N padded 56->64
  half_t* dtproj_h = (half_t*)take((size_t)2 * D_INNER * 32 * 2);      // K padded 24->32
  half_t* outw_h   = (half_t*)take((size_t)2 * D_MODEL * D_INNER * 2);
  // activations
  half_t* xn_h  = (half_t*)take((size_t)TOK * D_MODEL * 2);
  half_t* xl_h  = (half_t*)take((size_t)TOK * D_MODEL * 2);
  float*  z_f   = (float*) take((size_t)TOK * D_MODEL * 4);
  half_t* xb_h  = (half_t*)take((size_t)TOK * D_MODEL * 2);
  float*  xz_f  = (float*) take((size_t)TOK * (2 * D_INNER) * 4);
  float*  xc_f  = (float*) take((size_t)TOK * D_INNER * 4);
  half_t* xc_h  = (half_t*)take((size_t)TOK * D_INNER * 2);
  float*  dbc_f = (float*) take((size_t)TOK * 64 * 4);
  half_t* dtr_h = (half_t*)take((size_t)TOK * 32 * 2);
  float*  Bm_f  = (float*) take((size_t)TOK * D_STATE * 4);
  float*  Cm_f  = (float*) take((size_t)TOK * D_STATE * 4);
  float*  dt_f  = (float*) take((size_t)TOK * D_INNER * 4);
  half_t* y_h   = (half_t*)take((size_t)TOK * D_INNER * 2);
  float*  y0_f  = (float*) take((size_t)TOK * D_MODEL * 4);
  float*  y1_f  = (float*) take((size_t)TOK * D_MODEL * 4);

  auto cvt = [&](const float* s, half_t* d, int n) {
    cvt_f16_kernel<<<(n + 255) / 256, 256, 0, stream>>>(s, d, n);
  };
  auto gemm = [&](const half_t* A, const half_t* W, const float* bias,
                  float* oF, half_t* oH, int N, int K, int act) {
    int tasks = (TOK / 64) * (N / 16);
    int blocks = (tasks + 7) / 8;
    wmma_gemm<<<blocks, 256, 0, stream>>>(A, W, bias, oF, oH, N, K, act);
  };

  // --- weight conversion ---------------------------------------------------
  cvt(lin_w,   lin_h,   D_MODEL * D_MODEL);
  cvt(pconv_w, pconv_h, 2 * D_MODEL * D_MODEL);
  cvt(in_w,    in_h,    2 * (2 * D_INNER) * D_MODEL);
  cvt(out_w,   outw_h,  2 * D_MODEL * D_INNER);
  for (int i = 0; i < 2; ++i) {
    int np = 64 * D_INNER;
    cvt_pad_kernel<<<(np + 255) / 256, 256, 0, stream>>>(
        xproj_w + (size_t)i * (DT_RANK + 2 * D_STATE) * D_INNER,
        xproj_h + (size_t)i * np, DT_RANK + 2 * D_STATE, D_INNER, 64, D_INNER);
    int nd = D_INNER * 32;
    cvt_pad_kernel<<<(nd + 255) / 256, 256, 0, stream>>>(
        dtproj_w + (size_t)i * D_INNER * DT_RANK,
        dtproj_h + (size_t)i * nd, D_INNER, DT_RANK, D_INNER, 32);
  }

  // --- trunk ---------------------------------------------------------------
  ln_kernel<<<TOK, 128, 0, stream>>>(x, ln_g, ln_b, xn_h);
  gemm(xn_h, lin_h, lin_b, nullptr, xl_h, D_MODEL, D_MODEL, 0);          // xl
  gemm(xl_h, lin_h, lin_b, z_f,    nullptr, D_MODEL, D_MODEL, 1);       // z = silu(.)

  // --- two mamba branches --------------------------------------------------
  for (int i = 0; i < 2; ++i) {
    gemm(xl_h, pconv_h + (size_t)i * D_MODEL * D_MODEL,
         pconv_b + (size_t)i * D_MODEL, nullptr, xb_h, D_MODEL, D_MODEL, 1);
    gemm(xb_h, in_h + (size_t)i * (2 * D_INNER) * D_MODEL,
         nullptr, xz_f, nullptr, 2 * D_INNER, D_MODEL, 0);
    {
      int n = TOK * D_INNER;
      conv_silu_kernel<<<(n + 255) / 256, 256, 0, stream>>>(
          xz_f, conv_w + (size_t)i * D_INNER * D_CONV,
          conv_b + (size_t)i * D_INNER, xc_f, xc_h);
    }
    gemm(xc_h, xproj_h + (size_t)i * 64 * D_INNER, nullptr, dbc_f, nullptr, 64, D_INNER, 0);
    {
      int n = TOK * 64;
      pack_dbc_kernel<<<(n + 255) / 256, 256, 0, stream>>>(dbc_f, dtr_h, Bm_f, Cm_f);
    }
    gemm(dtr_h, dtproj_h + (size_t)i * D_INNER * 32,
         dtproj_b + (size_t)i * D_INNER, dt_f, nullptr, D_INNER, 32, 2); // softplus
    scan_kernel<<<BATCH * (D_INNER / 128), 128, 0, stream>>>(
        dt_f, xc_f, xz_f, Bm_f, Cm_f,
        A_log + (size_t)i * D_INNER * D_STATE, Dp + (size_t)i * D_INNER, y_h);
    gemm(y_h, outw_h + (size_t)i * D_MODEL * D_INNER, nullptr,
         (i ? y1_f : y0_f), nullptr, D_MODEL, D_INNER, 0);
  }

  // --- combine -------------------------------------------------------------
  {
    int n = TOK * D_MODEL;
    final_kernel<<<(n + 255) / 256, 256, 0, stream>>>(y0_f, y1_f, z_f, x, out, n);
  }
}
